// PlasticityModelMoE_64209761075279
// MI455X (gfx1250) — matrix operand order repaired and checked by reading, not compiled, not run on hardware
//
#include <hip/hip_runtime.h>
#include <hip/hip_bf16.h>

// Problem constants (from reference)
#define T_TOK 4096
#define D_DIM 512
#define F_DIM 2048
#define E_EXP 8
#define K_TOP 2

// MoE GEMM tiling
#define BM 32          // tokens per block tile
#define FN 256         // F columns per outer iteration
#define WS_STRIDE 40   // LDS stride (elements) for 32-wide K chunks of W1T/W2T (+8 pad)
#define XC_STRIDE 40   // LDS stride for X chunk
#define HS_STRIDE 264  // LDS stride for H tile (256 + 8 pad)

typedef __bf16 bf16;
typedef __attribute__((ext_vector_type(16))) __bf16 v16bf;
typedef __attribute__((ext_vector_type(8)))  __bf16 v8bf;
typedef __attribute__((ext_vector_type(8)))  float  v8f;

union ABFrag {
    v16bf v;
    struct { v8bf lo; v8bf hi; } h;
};

// ---- CDNA5 async global->LDS copies (ASYNCcnt-tracked, no VGPR round trip) ----
// VDST operand carries the 32-bit LDS byte offset; the low dword of a generic
// pointer into __shared__ is exactly that offset on AMDGPU. The instruction's
// INST_OFFSET immediate is added to BOTH the LDS and global addresses, so
// constant-stride bursts share one address pair with folded offsets.
template <int OFF>
__device__ __forceinline__ void async_copy_b128(void* lds_dst, const void* gsrc) {
    asm volatile("global_load_async_to_lds_b128 %0, %1, off offset:%2"
                 :: "v"((unsigned)(size_t)lds_dst), "v"(gsrc), "i"(OFF) : "memory");
}
__device__ __forceinline__ void async_copy_b64(void* lds_dst, const void* gsrc) {
    asm volatile("global_load_async_to_lds_b64 %0, %1, off"
                 :: "v"((unsigned)(size_t)lds_dst), "v"(gsrc) : "memory");
}
__device__ __forceinline__ void wait_async0() {
    asm volatile("s_wait_asynccnt 0x0" ::: "memory");
}

// ---------------- fp32 -> bf16 conversion ----------------
__global__ void k_conv_bf16(const float* __restrict__ src, bf16* __restrict__ dst, int n) {
    for (int i = blockIdx.x * blockDim.x + threadIdx.x; i < n; i += gridDim.x * blockDim.x)
        dst[i] = (bf16)src[i];
}

// src: [E][rows][cols] fp32  ->  dst: [E][cols][rows] bf16 (transposed per expert)
__global__ void k_conv_w_t(const float* __restrict__ src, bf16* __restrict__ dst,
                           int rows, int cols) {
    int per = rows * cols;
    int n = E_EXP * per;
    for (int i = blockIdx.x * blockDim.x + threadIdx.x; i < n; i += gridDim.x * blockDim.x) {
        int e = i / per;
        int rem = i - e * per;
        int r = rem / cols;
        int c = rem - r * cols;
        dst[(size_t)e * per + (size_t)c * rows + r] = (bf16)src[i];
    }
}

// ---------------- gating: logits -> top-2 softmax -> compacted lists ----------------
// 8 threads per token (one per expert). wave32: 4 tokens per wave.
__global__ void k_gate(const float* __restrict__ x, const float* __restrict__ Wg,
                       int* __restrict__ counts, int* __restrict__ tok,
                       float* __restrict__ gate) {
    int tid  = threadIdx.x;
    int lane = tid & 31;
    int grp  = tid >> 3;          // token index within block (0..31)
    int e    = tid & 7;           // expert for this thread
    int t    = blockIdx.x * 32 + grp;

    const float* xr = x + (size_t)t * D_DIM;
    float acc = 0.f;
    for (int d = 0; d < D_DIM; d += 4) {
        float4 xv = *(const float4*)(xr + d);
        acc += xv.x * Wg[(d + 0) * E_EXP + e];
        acc += xv.y * Wg[(d + 1) * E_EXP + e];
        acc += xv.z * Wg[(d + 2) * E_EXP + e];
        acc += xv.w * Wg[(d + 3) * E_EXP + e];
    }

    int base = lane & ~7;
    float v[E_EXP];
#pragma unroll
    for (int j = 0; j < E_EXP; ++j) v[j] = __shfl(acc, base + j, 32);

    if (e == 0) {
        int i0 = 0;
#pragma unroll
        for (int j = 1; j < E_EXP; ++j) if (v[j] > v[i0]) i0 = j;
        int i1 = (i0 == 0) ? 1 : 0;
#pragma unroll
        for (int j = 0; j < E_EXP; ++j) if (j != i0 && v[j] > v[i1]) i1 = j;
        // softmax over the two selected logits
        float p1 = __expf(v[i1] - v[i0]);
        float s  = 1.f + p1;
        float g0 = 1.f / s;
        float g1 = p1 / s;
        int p = atomicAdd(&counts[i0], 1);
        tok[i0 * T_TOK + p]  = t * 2 + 0;   // token*2 + slot
        gate[i0 * T_TOK + p] = g0;
        p = atomicAdd(&counts[i1], 1);
        tok[i1 * T_TOK + p]  = t * 2 + 1;
        gate[i1 * T_TOK + p] = g1;
    }
}

__device__ __forceinline__ float gelu_tanh(float v) {
    return 0.5f * v * (1.f + tanhf(0.7978845608028654f * (v + 0.044715f * v * v * v)));
}

// ---------------- fused expert MLP (grouped GEMM chain with WMMA bf16) ----------------
// Block = (expert e, 32-token tile). 8 waves:
//   wave = 2*wg + mw : mw in {0,1} selects 16-row half; wg in 0..3 selects column group.
// Phase 1: H[32 x 256] = gelu(X[32 x 512] @ W1 chunk + b1), streamed over K=512 in 32-steps.
// Phase 2: Y[32 x 512] += H[32 x 256] @ W2 chunk, streamed over K=256 in 32-steps.
// All global->LDS staging uses CDNA5 async-to-LDS instructions.
__launch_bounds__(256)
__global__ void k_moe(const bf16* __restrict__ Xb, const bf16* __restrict__ W1bT,
                      const bf16* __restrict__ W2bT, const float* __restrict__ b1,
                      const float* __restrict__ b2, const int* __restrict__ counts,
                      const int* __restrict__ tok, const float* __restrict__ gate,
                      float* __restrict__ ybuf) {
    __shared__ __align__(16) bf16 Ws[512 * WS_STRIDE];   // W1T (256 rows used) or W2T (512 rows) K-chunk
    __shared__ __align__(16) bf16 Hs[BM * HS_STRIDE];    // H tile, bf16
    __shared__ __align__(16) bf16 Xc[BM * XC_STRIDE];    // X K-chunk
    __shared__ int   s_tok[BM];
    __shared__ float s_gate[BM];

    const int ntiles = T_TOK / BM;        // 128
    int e        = blockIdx.x / ntiles;
    int tile     = blockIdx.x - e * ntiles;
    int cnt      = counts[e];
    int row_base = tile * BM;
    if (row_base >= cnt) return;          // empty tile: whole block exits
    int m_count  = min(BM, cnt - row_base);

    int tid  = threadIdx.x;
    int lane = tid & 31;
    int wave = tid >> 5;
    int mw   = wave & 1;                  // which 16-row half
    int wg   = wave >> 1;                 // column group 0..3
    const int half8 = (lane < 16) ? 0 : 8;
    const int nlan  = lane & 15;
    const int kofs  = (lane < 16) ? 0 : 16;

    if (tid < BM) {
        if (tid < m_count) {
            s_tok[tid]  = tok[e * T_TOK + row_base + tid];
            s_gate[tid] = gate[e * T_TOK + row_base + tid];
        } else {
            s_tok[tid]  = -1;
            s_gate[tid] = 0.f;
        }
    }
    __syncthreads();

    // Zero-fill X rows of the padded tile once; async loads never touch them.
    {
        int m = tid >> 3, sub = tid & 7;
        if (s_tok[m] < 0)
            *((uint2*)(Xc + m * XC_STRIDE) + sub) = make_uint2(0u, 0u);
    }

    const bf16* W1e = W1bT + (size_t)e * F_DIM * D_DIM;  // [F][D]
    const bf16* W2e = W2bT + (size_t)e * D_DIM * F_DIM;  // [D][F]

    v8f yacc[8] = {};                     // 8 d-tiles of 16x16 per wave -> 64 VGPRs

    for (int fc = 0; fc < F_DIM; fc += FN) {
        v8f hacc[4] = {};                 // 4 f-tiles of 16x16 per wave

        // ---------- Phase 1: H = gelu(X @ W1 chunk + b1) ----------
        for (int k0 = 0; k0 < D_DIM; k0 += 32) {
            __syncthreads();              // previous consumers of Ws/Xc are done
            // W1T chunk: 256 f-rows x 32 d-cols; one 64B row per thread
            // (4x b128 async, folded offset immediates on one address pair)
            {
                const bf16* src = W1e + (size_t)(fc + tid) * D_DIM + k0;
                bf16* dst = Ws + tid * WS_STRIDE;
                async_copy_b128<0>(dst, src);
                async_copy_b128<16>(dst, src);
                async_copy_b128<32>(dst, src);
                async_copy_b128<48>(dst, src);
            }
            // X chunk: 32 token-rows x 32 d-cols; 8 threads per row, 8B each (b64 async)
            {
                int m = tid >> 3, sub = tid & 7;
                int tk = s_tok[m];
                if (tk >= 0)
                    async_copy_b64(Xc + m * XC_STRIDE + sub * 4,
                                   Xb + (size_t)(tk >> 1) * D_DIM + k0 + sub * 4);
            }
            wait_async0();
            __syncthreads();              // all waves' async data visible in LDS

            // A fragment: row M = mw*16 + nlan; K segments per documented 16-bit A layout
            ABFrag a;
            const bf16* ap = Xc + (mw * 16 + nlan) * XC_STRIDE;
            a.h.lo = *(const v8bf*)(ap + half8);
            a.h.hi = *(const v8bf*)(ap + half8 + 16);
#pragma unroll
            for (int j = 0; j < 4; ++j) {
                // B fragment: column N = nlan of B (32x16); contiguous in transposed LDS
                ABFrag b;
                const bf16* bp = Ws + (wg * 64 + j * 16 + nlan) * WS_STRIDE + half8;
                b.h.lo = *(const v8bf*)(bp);
                b.h.hi = *(const v8bf*)(bp + 16);
                hacc[j] = __builtin_amdgcn_wmma_f32_16x16x32_bf16(
                    false, a.v, false, b.v, (short)0, hacc[j], false, false);
            }
        }

        // epilogue: +b1, gelu, store H tile to LDS as bf16 (row-major for phase-2 A frags)
#pragma unroll
        for (int j = 0; j < 4; ++j) {
            int fl = wg * 64 + j * 16 + nlan;
            float bias = b1[e * F_DIM + fc + fl];
#pragma unroll
            for (int r = 0; r < 8; ++r) {
                float g = gelu_tanh(hacc[j][r] + bias);
                Hs[(mw * 16 + r + half8) * HS_STRIDE + fl] = (bf16)g;
            }
        }

        // ---------- Phase 2: Y += H @ W2 chunk ----------
        for (int k0 = 0; k0 < FN; k0 += 32) {
            __syncthreads();              // covers Hs writes and Ws reuse
            // W2T chunk: 512 d-rows x 32 f-cols; two 64B rows per thread
#pragma unroll
            for (int rr = 0; rr < 2; ++rr) {
                int d = tid + rr * 256;
                const bf16* src = W2e + (size_t)d * F_DIM + fc + k0;
                bf16* dst = Ws + d * WS_STRIDE;
                async_copy_b128<0>(dst, src);
                async_copy_b128<16>(dst, src);
                async_copy_b128<32>(dst, src);
                async_copy_b128<48>(dst, src);
            }
            wait_async0();
            __syncthreads();

            ABFrag a;
            const bf16* ap = Hs + (mw * 16 + nlan) * HS_STRIDE + k0;
            a.h.lo = *(const v8bf*)(ap + half8);
            a.h.hi = *(const v8bf*)(ap + half8 + 16);
#pragma unroll
            for (int j = 0; j < 8; ++j) {
                ABFrag b;
                const bf16* bp = Ws + (wg * 128 + j * 16 + nlan) * WS_STRIDE + half8;
                b.h.lo = *(const v8bf*)(bp);
                b.h.hi = *(const v8bf*)(bp + 16);
                yacc[j] = __builtin_amdgcn_wmma_f32_16x16x32_bf16(
                    false, a.v, false, b.v, (short)0, yacc[j], false, false);
            }
        }
    }

    // ---------- output: ybuf[slot][t][d] = gate * (y + b2) ----------
    int tks[8]; float gts[8];
#pragma unroll
    for (int r = 0; r < 8; ++r) {
        int m = mw * 16 + r + half8;
        tks[r] = s_tok[m];
        gts[r] = s_gate[m];
    }
#pragma unroll
    for (int j = 0; j < 8; ++j) {
        int d = wg * 128 + j * 16 + nlan;
        float b2v = b2[e * D_DIM + d];
#pragma unroll
        for (int r = 0; r < 8; ++r) {
            if (tks[r] >= 0) {
                int t = tks[r] >> 1, slot = tks[r] & 1;
                ybuf[(size_t)slot * T_TOK * D_DIM + (size_t)t * D_DIM + d] =
                    gts[r] * (yacc[j][r] + b2v);
            }
        }
    }
}

// out = ybuf[slot 0] + ybuf[slot 1]
__global__ void k_combine(const float* __restrict__ ybuf, float* __restrict__ out, int n) {
    for (int i = blockIdx.x * blockDim.x + threadIdx.x; i < n; i += gridDim.x * blockDim.x)
        out[i] = ybuf[i] + ybuf[(size_t)n + i];
}

extern "C" void kernel_launch(void* const* d_in, const int* in_sizes, int n_in,
                              void* d_out, int out_size, void* d_ws, size_t ws_size,
                              hipStream_t stream) {
    const float* x  = (const float*)d_in[0];
    const float* Wg = (const float*)d_in[1];
    const float* W1 = (const float*)d_in[2];
    const float* b1 = (const float*)d_in[3];
    const float* W2 = (const float*)d_in[4];
    const float* b2 = (const float*)d_in[5];
    float* out = (float*)d_out;

    char* ws = (char*)d_ws;
    size_t off = 0;
    auto wsa = [&](size_t bytes) -> void* {
        void* p = ws + off;
        off = (off + bytes + 255) & ~(size_t)255;
        return p;
    };
    int*   counts = (int*)  wsa(E_EXP * sizeof(int));
    int*   tok    = (int*)  wsa((size_t)E_EXP * T_TOK * sizeof(int));
    float* gate   = (float*)wsa((size_t)E_EXP * T_TOK * sizeof(float));
    bf16*  Xb     = (bf16*) wsa((size_t)T_TOK * D_DIM * sizeof(bf16));
    bf16*  W1bT   = (bf16*) wsa((size_t)E_EXP * D_DIM * F_DIM * sizeof(bf16));
    bf16*  W2bT   = (bf16*) wsa((size_t)E_EXP * D_DIM * F_DIM * sizeof(bf16));
    float* ybuf   = (float*)wsa((size_t)K_TOP * T_TOK * D_DIM * sizeof(float));

    hipMemsetAsync(counts, 0, E_EXP * sizeof(int), stream);

    k_conv_bf16<<<2048, 256, 0, stream>>>(x, Xb, T_TOK * D_DIM);
    k_conv_w_t<<<8192, 256, 0, stream>>>(W1, W1bT, D_DIM, F_DIM);   // -> [E][F][D]
    k_conv_w_t<<<8192, 256, 0, stream>>>(W2, W2bT, F_DIM, D_DIM);   // -> [E][D][F]

    k_gate<<<T_TOK / 32, 256, 0, stream>>>(x, Wg, counts, tok, gate);

    k_moe<<<E_EXP * (T_TOK / BM), 256, 0, stream>>>(Xb, W1bT, W2bT, b1, b2,
                                                    counts, tok, gate, ybuf);

    k_combine<<<4096, 256, 0, stream>>>(ybuf, out, T_TOK * D_DIM);
}
